// JointAttention_49950469653327
// MI455X (gfx1250) — matrix operand order repaired and checked by reading, not compiled, not run on hardware
//
#include <hip/hip_runtime.h>
#include <hip/hip_bf16.h>

// ---------------------------------------------------------------------------
// Types for CDNA5 WMMA (wave32, 16x16x32 bf16 -> f32) and TDM descriptors
// ---------------------------------------------------------------------------
typedef __attribute__((ext_vector_type(16))) __bf16 v16bf;
typedef __attribute__((ext_vector_type(8)))  float  v8f;
typedef __attribute__((ext_vector_type(4)))  unsigned int v4u;
typedef __attribute__((ext_vector_type(4)))  int    v4i;
typedef __attribute__((ext_vector_type(8)))  int    v8i;

union FragA {                 // one wave-lane's share of a 16x32 bf16 A/B tile
  v16bf          bf;
  uint4          q[2];
  unsigned short us[16];
  unsigned int   u[8];
};

static __device__ inline unsigned short f2bf(float f) {
  unsigned int u = __float_as_uint(f);
  unsigned int r = u + 0x7FFFu + ((u >> 16) & 1u);   // round-to-nearest-even
  return (unsigned short)(r >> 16);
}
static __device__ inline float bf2f(unsigned short h) {
  return __uint_as_float(((unsigned int)h) << 16);
}
static __device__ inline v8f vzero8() {
  v8f z = {0.f, 0.f, 0.f, 0.f, 0.f, 0.f, 0.f, 0.f};
  return z;
}

#define DIMX     2048
#define SLEN     2048
#define NH       16
#define NKV      8
#define HD       128
#define QKV_OUT  4096

#if __has_builtin(__builtin_amdgcn_tensor_load_to_lds)
#define HAVE_TDM 1
#else
#define HAVE_TDM 0
#endif

#if HAVE_TDM
// ---------------------------------------------------------------------------
// Tensor Data Mover: DMA a 2D bf16 tile [tile_d1 rows x tile_d0 elems] from
// global (row stride = stride0 elements) into LDS at byte offset lds_off.
// D# layout per CDNA5 ISA (count=1, type=2, 2B elems,
// workgroup_mask=0: not in a cluster).
// ---------------------------------------------------------------------------
static __device__ inline void tdm_load_2d(unsigned lds_off, const void* gaddr,
                                          unsigned tile_d0, unsigned tile_d1,
                                          unsigned tensor_d0, unsigned tensor_d1,
                                          unsigned stride0_elems) {
  unsigned long long ga = (unsigned long long)(size_t)gaddr;
  v4u g0;
  g0[0] = 1u;                                            // count=1, user mode
  g0[1] = lds_off;                                       // lds_addr (bytes)
  g0[2] = (unsigned)(ga & 0xFFFFFFFFull);                // global_addr[31:0]
  g0[3] = (unsigned)((ga >> 32) & 0x1FFFFFFull)          // global_addr[56:32]
        | (2u << 30);                                    // type=2 ("image")
  v8i g1;
  g1[0] = (int)(1u << 16);                               // data_size=1 (2B), mask=0
  g1[1] = (int)((tensor_d0 & 0xFFFFu) << 16);            // tensor_dim0[15:0]
  g1[2] = (int)((tensor_d0 >> 16) | ((tensor_d1 & 0xFFFFu) << 16));
  g1[3] = (int)((tensor_d1 >> 16) | (tile_d0 << 16));    // tile_dim0
  g1[4] = (int)(tile_d1 & 0xFFFFu);                      // tile_dim1, tile_dim2=0
  g1[5] = (int)stride0_elems;                            // tensor_dim0_stride[31:0]
  g1[6] = 0;
  g1[7] = 0;
  v4i g2 = {0, 0, 0, 0};
  v4i g3 = {0, 0, 0, 0};
#if __clang_major__ >= 23
  v8i g4 = {0, 0, 0, 0, 0, 0, 0, 0};
  __builtin_amdgcn_tensor_load_to_lds(g0, g1, g2, g3, g4, 0);
#else
  __builtin_amdgcn_tensor_load_to_lds(g0, g1, g2, g3, 0);
#endif
}
#endif  // HAVE_TDM

// ---------------------------------------------------------------------------
// f32 -> bf16 conversion (grid-stride)
// ---------------------------------------------------------------------------
__global__ void cvt_f32_bf16(const float* __restrict__ in,
                             unsigned short* __restrict__ out, long long n) {
  long long i = (long long)blockIdx.x * blockDim.x + threadIdx.x;
  long long stride = (long long)gridDim.x * blockDim.x;
  for (; i < n; i += stride) out[i] = f2bf(in[i]);
}

// ---------------------------------------------------------------------------
// GEMM: C[M,N] = A[M,K] * B[N,K]^T, A/B bf16 row-major (K contiguous).
// Block = 256 threads = 8 waves; block tile 128x128; wave tile 16x128.
// K-slices (128x32 bf16, 8KB) of A and B are staged in LDS by the Tensor
// Data Mover (double buffered, s_wait_tensorcnt + barrier), all waves read
// fragments with ds_load_b128.
// ---------------------------------------------------------------------------
template <bool OUT_BF16>
__global__ __launch_bounds__(256)
void gemm_bf16_wmma(const unsigned short* __restrict__ A,
                    const unsigned short* __restrict__ B,
                    void* __restrict__ Cout, int M, int N, int K) {
  const int lane  = threadIdx.x & 31;
  const int wave  = threadIdx.x >> 5;
  const int row0  = blockIdx.x * 128;
  const int col0  = blockIdx.y * 128;
  const int ncol  = lane & 15;
  const int hi    = lane >> 4;
  const int kbase = hi ? 8 : 0;      // A-frag K start per lane half
  const int khalf = hi ? 8 : 0;      // B-frag K start per lane half

  v8f acc[8];
#pragma unroll
  for (int t = 0; t < 8; ++t) acc[t] = vzero8();

#if HAVE_TDM
  __shared__ unsigned short Atile[2][128][32];
  __shared__ unsigned short Btile[2][128][32];

  if (wave == 0) {
    tdm_load_2d((unsigned)(size_t)&Atile[0][0][0], A + (long long)row0 * K,
                32u, 128u, (unsigned)K, (unsigned)M, (unsigned)K);
    tdm_load_2d((unsigned)(size_t)&Btile[0][0][0], B + (long long)col0 * K,
                32u, 128u, (unsigned)K, (unsigned)N, (unsigned)K);
  }

  int cur = 0;
  for (int k0 = 0; k0 < K; k0 += 32) {
    const bool more = (k0 + 32) < K;
    if (wave == 0) {
      if (more) {
        tdm_load_2d((unsigned)(size_t)&Atile[cur ^ 1][0][0],
                    A + (long long)row0 * K + (k0 + 32),
                    32u, 128u, (unsigned)K, (unsigned)M, (unsigned)K);
        tdm_load_2d((unsigned)(size_t)&Btile[cur ^ 1][0][0],
                    B + (long long)col0 * K + (k0 + 32),
                    32u, 128u, (unsigned)K, (unsigned)N, (unsigned)K);
        __builtin_amdgcn_s_wait_tensorcnt(2);  // current pair has landed
      } else {
        __builtin_amdgcn_s_wait_tensorcnt(0);
      }
    }
    __syncthreads();   // publish current LDS tiles to all waves

    FragA a;
    {
      const unsigned short* pa = &Atile[cur][wave * 16 + ncol][kbase];
      a.q[0] = *(const uint4*)pa;
      a.q[1] = *(const uint4*)(pa + 16);
    }
#pragma unroll
    for (int t = 0; t < 8; ++t) {
      FragA bfr;
      const unsigned short* pb = &Btile[cur][t * 16 + ncol][khalf];
      bfr.q[0] = *(const uint4*)pb;
      bfr.q[1] = *(const uint4*)(pb + 16);
      acc[t] = __builtin_amdgcn_wmma_f32_16x16x32_bf16(
          false, a.bf, false, bfr.bf, (short)0, acc[t], false, false);
    }
    __syncthreads();   // all reads done before this buffer is overwritten
    cur ^= 1;
  }
#else
  // Fallback: direct global fragment loads (no TDM builtin available)
  const int rowA = row0 + wave * 16 + ncol;
  for (int k0 = 0; k0 < K; k0 += 32) {
    FragA a;
    const unsigned short* pa = A + (long long)rowA * K + k0 + kbase;
    a.q[0] = *(const uint4*)pa;
    a.q[1] = *(const uint4*)(pa + 16);
#pragma unroll
    for (int t = 0; t < 8; ++t) {
      FragA bfr;
      const unsigned short* pb =
          B + (long long)(col0 + t * 16 + ncol) * K + k0 + khalf;
      bfr.q[0] = *(const uint4*)pb;
      bfr.q[1] = *(const uint4*)(pb + 16);
      acc[t] = __builtin_amdgcn_wmma_f32_16x16x32_bf16(
          false, a.bf, false, bfr.bf, (short)0, acc[t], false, false);
    }
  }
#endif

  const int mrow = hi * 8;
  const int rbase = row0 + wave * 16 + mrow;
  if (OUT_BF16) {
    unsigned short* C = (unsigned short*)Cout;
#pragma unroll
    for (int t = 0; t < 8; ++t)
#pragma unroll
      for (int j = 0; j < 8; ++j)
        C[(long long)(rbase + j) * N + col0 + t * 16 + ncol] = f2bf(acc[t][j]);
  } else {
    float* C = (float*)Cout;
#pragma unroll
    for (int t = 0; t < 8; ++t)
#pragma unroll
      for (int j = 0; j < 8; ++j)
        C[(long long)(rbase + j) * N + col0 + t * 16 + ncol] = acc[t][j];
  }
}

// ---------------------------------------------------------------------------
// QKV post-processing: RMSNorm(q,k) + rotary, repack into attention layouts.
//  Qb: [B, NH,  S, HD]   (row = query, K-contig head dim)   -> A frags
//  Kb: [B, NKV, S, HD]   (row = key,   K-contig head dim)   -> B frags (Q*K^T)
//  Vt: [B, NKV, HD, S]   (row = dim,   S-contig)            -> B frags (P*V)
// ---------------------------------------------------------------------------
__global__ __launch_bounds__(256)
void qkv_post_kernel(const unsigned short* __restrict__ qkv,  // [B*S, 4096]
                     const float* __restrict__ freqs,         // [S,1,64,2,2]
                     const float* __restrict__ q_w,
                     const float* __restrict__ k_w,
                     unsigned short* __restrict__ Qb,
                     unsigned short* __restrict__ Kb,
                     unsigned short* __restrict__ Vt) {
  const int token = blockIdx.x;               // b*S + s
  const int b = token / SLEN, s = token - b * SLEN;
  const int t = threadIdx.x;
  const int head = t >> 3;                    // 0..31 (16 q, 8 k, 8 v)
  const int part = t & 7;
  const int d0 = part * 16;

  const unsigned short* src = qkv + (long long)token * QKV_OUT + head * HD + d0;
  float x[16];
#pragma unroll
  for (int i = 0; i < 16; ++i) x[i] = bf2f(src[i]);

  if (head < 24) {  // q or k: rmsnorm + rotary
    float ss = 0.f;
#pragma unroll
    for (int i = 0; i < 16; ++i) ss += x[i] * x[i];
    ss += __shfl_xor(ss, 1);
    ss += __shfl_xor(ss, 2);
    ss += __shfl_xor(ss, 4);
    float rms = rsqrtf(ss * (1.0f / 128.0f) + 1e-5f);
    const float* w = (head < NH) ? q_w : k_w;
#pragma unroll
    for (int i = 0; i < 16; ++i) x[i] = x[i] * rms * w[d0 + i];
    const float* f = freqs + (long long)s * 256 + d0 * 2;
    float y[16];
#pragma unroll
    for (int u = 0; u < 8; ++u) {
      float x0 = x[2 * u], x1 = x[2 * u + 1];
      const float* fp = f + u * 4;
      y[2 * u]     = fp[0] * x0 + fp[1] * x1;
      y[2 * u + 1] = fp[2] * x0 + fp[3] * x1;
    }
#pragma unroll
    for (int i = 0; i < 16; ++i) x[i] = y[i];
  }

  if (head < NH) {
    unsigned short* dst =
        Qb + (((long long)(b * NH + head)) * SLEN + s) * HD + d0;
#pragma unroll
    for (int i = 0; i < 16; ++i) dst[i] = f2bf(x[i]);
  } else if (head < 24) {
    int kh = head - NH;
    unsigned short* dst =
        Kb + (((long long)(b * NKV + kh)) * SLEN + s) * HD + d0;
#pragma unroll
    for (int i = 0; i < 16; ++i) dst[i] = f2bf(x[i]);
  } else {
    int vh = head - 24;
    unsigned short* base =
        Vt + ((long long)(b * NKV + vh) * HD + d0) * SLEN + s;
#pragma unroll
    for (int i = 0; i < 16; ++i) base[(long long)i * SLEN] = f2bf(x[i]);
  }
}

// ---------------------------------------------------------------------------
// Flash attention: grid (S/128, B*NH); 8 waves/block; wave owns 16 q-rows.
// Streams K/V in 32-key tiles; scores + PV both via bf16 WMMA; P re-laid
// out C-frag -> A-frag through LDS.
// ---------------------------------------------------------------------------
__global__ __launch_bounds__(256)
void flash_attn_kernel(const unsigned short* __restrict__ Qb,
                       const unsigned short* __restrict__ Kb,
                       const unsigned short* __restrict__ Vt,
                       const float* __restrict__ mask,   // [B,1,S,S]
                       unsigned short* __restrict__ Ob)  // [B*S, 2048]
{
  __shared__ float pbuf[8][16][33];   // per-wave P tile, padded rows
  const int lane  = threadIdx.x & 31;
  const int wave  = threadIdx.x >> 5;
  const int bh    = blockIdx.y;       // b*NH + h
  const int b     = bh >> 4, h = bh & 15;
  const int kvh   = h >> 1;           // GQA: consecutive replication
  const int q0    = blockIdx.x * 128 + wave * 16;
  const int ncol  = lane & 15;
  const int hi    = lane >> 4;
  const int kbase = hi ? 8 : 0;
  const int khalf = hi ? 8 : 0;
  const float scale = 0.08838834764831845f;  // 1/sqrt(128)

  FragA qf[4];
  {
    const unsigned short* pq = Qb + (((long long)bh) * SLEN + q0 + ncol) * HD;
#pragma unroll
    for (int c = 0; c < 4; ++c) {
      const unsigned short* p = pq + c * 32 + kbase;
      qf[c].q[0] = *(const uint4*)p;
      qf[c].q[1] = *(const uint4*)(p + 16);
    }
  }

  v8f acc[8];
#pragma unroll
  for (int t = 0; t < 8; ++t) acc[t] = vzero8();
  float mrow[8], lrow[8];
#pragma unroll
  for (int j = 0; j < 8; ++j) { mrow[j] = -1e30f; lrow[j] = 0.f; }

  const unsigned short* Kbase = Kb + (long long)(b * NKV + kvh) * SLEN * HD;
  const unsigned short* Vbase = Vt + (long long)(b * NKV + kvh) * HD * SLEN;
  const float* mbase = mask + ((long long)b * SLEN + q0) * SLEN;

  for (int j0 = 0; j0 < SLEN; j0 += 32) {
    v8f c0 = vzero8(), c1 = vzero8();
#pragma unroll
    for (int c = 0; c < 4; ++c) {
      FragA kf0, kf1;
      const unsigned short* p0 =
          Kbase + (long long)(j0 + ncol) * HD + c * 32 + khalf;
      const unsigned short* p1 =
          Kbase + (long long)(j0 + 16 + ncol) * HD + c * 32 + khalf;
      kf0.q[0] = *(const uint4*)p0; kf0.q[1] = *(const uint4*)(p0 + 16);
      kf1.q[0] = *(const uint4*)p1; kf1.q[1] = *(const uint4*)(p1 + 16);
      c0 = __builtin_amdgcn_wmma_f32_16x16x32_bf16(
          false, qf[c].bf, false, kf0.bf, (short)0, c0, false, false);
      c1 = __builtin_amdgcn_wmma_f32_16x16x32_bf16(
          false, qf[c].bf, false, kf1.bf, (short)0, c1, false, false);
    }
    __builtin_prefetch(Kbase + (long long)(j0 + 32 + ncol) * HD, 0, 0);

#pragma unroll
    for (int j = 0; j < 8; ++j) {
      const int m = j + 8 * hi;
      c0[j] = c0[j] * scale + mbase[(long long)m * SLEN + j0 + ncol];
      c1[j] = c1[j] * scale + mbase[(long long)m * SLEN + j0 + 16 + ncol];
      float v = fmaxf(c0[j], c1[j]);
      v = fmaxf(v, __shfl_xor(v, 1));
      v = fmaxf(v, __shfl_xor(v, 2));
      v = fmaxf(v, __shfl_xor(v, 4));
      v = fmaxf(v, __shfl_xor(v, 8));
      const float mnew = fmaxf(mrow[j], v);
      const float corr = __expf(mrow[j] - mnew);
      c0[j] = __expf(c0[j] - mnew);
      c1[j] = __expf(c1[j] - mnew);
      float rs = c0[j] + c1[j];
      rs += __shfl_xor(rs, 1);
      rs += __shfl_xor(rs, 2);
      rs += __shfl_xor(rs, 4);
      rs += __shfl_xor(rs, 8);
      lrow[j] = lrow[j] * corr + rs;
      mrow[j] = mnew;
#pragma unroll
      for (int t = 0; t < 8; ++t) acc[t][j] *= corr;
    }

    // P: C-frag layout -> A-frag layout via per-wave LDS bounce
#pragma unroll
    for (int j = 0; j < 8; ++j) {
      const int m = j + 8 * hi;
      pbuf[wave][m][ncol]      = c0[j];
      pbuf[wave][m][16 + ncol] = c1[j];
    }
    __builtin_amdgcn_wave_barrier();
    FragA pf;
#pragma unroll
    for (int i = 0; i < 8; ++i) {
      pf.us[i]     = f2bf(pbuf[wave][ncol][kbase + i]);
      pf.us[i + 8] = f2bf(pbuf[wave][ncol][kbase + 16 + i]);
    }
    __builtin_amdgcn_wave_barrier();

#pragma unroll
    for (int t = 0; t < 8; ++t) {
      FragA vf;
      const unsigned short* p =
          Vbase + (long long)(t * 16 + ncol) * SLEN + j0 + khalf;
      vf.q[0] = *(const uint4*)p;
      vf.q[1] = *(const uint4*)(p + 16);
      acc[t] = __builtin_amdgcn_wmma_f32_16x16x32_bf16(
          false, pf.bf, false, vf.bf, (short)0, acc[t], false, false);
    }
  }

#pragma unroll
  for (int j = 0; j < 8; ++j) {
    const float inv = 1.0f / lrow[j];
    const int m = j + 8 * hi;
    unsigned short* dst =
        Ob + ((long long)(b * SLEN + q0 + m)) * DIMX + h * HD;
#pragma unroll
    for (int t = 0; t < 8; ++t) dst[t * 16 + ncol] = f2bf(acc[t][j] * inv);
  }
}

// ---------------------------------------------------------------------------
// Host-side launch
// ---------------------------------------------------------------------------
extern "C" void kernel_launch(void* const* d_in, const int* in_sizes, int n_in,
                              void* d_out, int out_size, void* d_ws,
                              size_t ws_size, hipStream_t stream) {
  (void)in_sizes; (void)n_in; (void)out_size; (void)ws_size;
  const float* x     = (const float*)d_in[0];  // [B,S,DIM]
  const float* xmask = (const float*)d_in[1];  // [B,1,S,S]
  const float* freqs = (const float*)d_in[2];  // [S,1,64,2,2]
  const float* w_qkv = (const float*)d_in[3];  // [4096,2048]
  const float* w_out = (const float*)d_in[4];  // [2048,2048]
  const float* q_w   = (const float*)d_in[5];  // [128]
  const float* k_w   = (const float*)d_in[6];  // [128]
  float* out = (float*)d_out;

  const long long M  = 2LL * SLEN;        // 4096 tokens
  char* ws = (char*)d_ws;
  unsigned short* xb    = (unsigned short*)(ws);                    // 16 MB
  unsigned short* wqb   = (unsigned short*)(ws + (16LL << 20));     // 16 MB
  unsigned short* wob   = (unsigned short*)(ws + (32LL << 20));     //  8 MB
  unsigned short* qkvb  = (unsigned short*)(ws + (40LL << 20));     // 32 MB
  unsigned short* Qb    = (unsigned short*)(ws + (72LL << 20));     // 16 MB
  unsigned short* Kb    = (unsigned short*)(ws + (88LL << 20));     //  8 MB
  unsigned short* Vt    = (unsigned short*)(ws + (96LL << 20));     //  8 MB
  unsigned short* attb  = (unsigned short*)(ws + (104LL << 20));    // 16 MB

  cvt_f32_bf16<<<1024, 256, 0, stream>>>(x, xb, M * DIMX);
  cvt_f32_bf16<<<1024, 256, 0, stream>>>(w_qkv, wqb, (long long)QKV_OUT * DIMX);
  cvt_f32_bf16<<<1024, 256, 0, stream>>>(w_out, wob, (long long)DIMX * DIMX);

  gemm_bf16_wmma<true><<<dim3(32, 32), 256, 0, stream>>>(
      xb, wqb, qkvb, (int)M, QKV_OUT, DIMX);

  qkv_post_kernel<<<(int)M, 256, 0, stream>>>(qkvb, freqs, q_w, k_w, Qb, Kb, Vt);

  flash_attn_kernel<<<dim3(SLEN / 128, 2 * NH), 256, 0, stream>>>(
      Qb, Kb, Vt, xmask, attb);

  gemm_bf16_wmma<false><<<dim3(32, 16), 256, 0, stream>>>(
      attb, wob, out, (int)M, DIMX, DIMX);
}
